// S5SSM_37658273252076
// MI455X (gfx1250) — compile-verified, matrix operand order checked by reading, not run on hardware
//
#include <hip/hip_runtime.h>
#include <hip/hip_bf16.h>
#include <math.h>

#define BSZd    4
#define LSEQ    4096
#define HDIM    1024
#define PDIM    512
#define P2      1024          // 2*PDIM (re | im concatenated along N/K)
#define CHUNK_M 128           // GEMM row tile per workgroup
#define SCL     256           // scan chunk length (= 2^8)
#define NCH     (LSEQ / SCL)  // 16

typedef __bf16 bf16_t;
typedef __attribute__((ext_vector_type(8)))  __bf16 bfx8;
typedef __attribute__((ext_vector_type(16))) __bf16 bfx16;
typedef __attribute__((ext_vector_type(8)))  float  f32x8;

union ABFrag { bfx16 v; bfx8 h[2]; };

// ---------------------------------------------------------------------------
// K0a: per-p parameters + B_bar (bf16, [2P][H], rows 0..511 = re, 512..1023 = im)
// ---------------------------------------------------------------------------
__global__ void s5_prep_params(const float* __restrict__ Lre, const float* __restrict__ Lim,
                               const float* __restrict__ Bmat, const float* __restrict__ log_step,
                               unsigned short* __restrict__ Bbar_u16,
                               float* __restrict__ lam_r, float* __restrict__ lam_i,
                               float* __restrict__ dscl_r, float* __restrict__ dscl_i) {
  const int p = threadIdx.x;
  if (p >= PDIM) return;
  bf16_t* Bbar = (bf16_t*)Bbar_u16;
  const float lr = Lre[p], li = Lim[p];
  const float st = expf(log_step[p]);
  const float er = expf(lr * st);
  const float ang = li * st;
  const float lbr = er * cosf(ang);
  const float lbi = er * sinf(ang);
  lam_r[p] = lbr; lam_i[p] = lbi;
  // lambda_bar^SCL via 8 complex squarings (SCL = 256)
  float dr = lbr, di = lbi;
#pragma unroll
  for (int s = 0; s < 8; ++s) { float nr = dr*dr - di*di; float ni = 2.f*dr*di; dr = nr; di = ni; }
  dscl_r[p] = dr; dscl_i[p] = di;
  // B_scale = (lambda_bar - 1) / Lambda  (complex divide)
  const float den = lr*lr + li*li;
  const float nr = lbr - 1.f, ni = lbi;
  const float bsr = (nr*lr + ni*li) / den;
  const float bsi = (ni*lr - nr*li) / den;
  for (int h = 0; h < HDIM; ++h) {
    const float b0 = Bmat[((size_t)p*HDIM + h)*2 + 0];
    const float b1 = Bmat[((size_t)p*HDIM + h)*2 + 1];
    Bbar[(size_t)p*HDIM + h]          = (bf16_t)(bsr*b0 - bsi*b1);
    Bbar[(size_t)(p+PDIM)*HDIM + h]   = (bf16_t)(bsr*b1 + bsi*b0);
  }
}

// ---------------------------------------------------------------------------
// K0b: Ccat bf16, [H][2P] : cols 0..511 = Cr, 512..1023 = Ci
// ---------------------------------------------------------------------------
__global__ void s5_prep_c(const float* __restrict__ Cmat, unsigned short* __restrict__ Ccat_u16) {
  bf16_t* Ccat = (bf16_t*)Ccat_u16;
  const int i = blockIdx.x * blockDim.x + threadIdx.x;
  if (i >= HDIM * PDIM) return;
  const int h = i / PDIM, p = i % PDIM;
  Ccat[(size_t)h*P2 + p]        = (bf16_t)Cmat[((size_t)h*PDIM + p)*2 + 0];
  Ccat[(size_t)h*P2 + p + PDIM] = (bf16_t)Cmat[((size_t)h*PDIM + p)*2 + 1];
}

// ---------------------------------------------------------------------------
// K1: Bu[b][l][n] = sum_h u[b][l][h] * Bbar[n][h]   (bf16 WMMA, fp32 acc)
//     u is stored [b][h][l]; a 128(l) x 1024(h) bf16 tile is staged in LDS.
// ---------------------------------------------------------------------------
__global__ void __launch_bounds__(256) s5_gemm_bu(const float* __restrict__ u,
    const unsigned short* __restrict__ Bbar_u16, float* __restrict__ Bu) {
  extern __shared__ __align__(16) unsigned short smem_raw[];
  bf16_t* sU = (bf16_t*)smem_raw;                 // [CHUNK_M][HDIM+8]
  const int LDK = HDIM + 8;
  const bf16_t* Bbar = (const bf16_t*)Bbar_u16;
  const int tid = threadIdx.x, lane = tid & 31, wv = tid >> 5;
  const int b = blockIdx.y;
  const int l0 = blockIdx.x * CHUNK_M;

  // stage: transpose [h][l] -> LDS [l][h], convert fp32 -> bf16
  for (int h = wv; h < HDIM; h += 8) {
    const float4 f = *(const float4*)(u + ((size_t)b*HDIM + h)*LSEQ + l0 + lane*4);
    bf16_t* d = sU + (size_t)(lane*4)*LDK + h;
    d[0]     = (bf16_t)f.x;
    d[LDK]   = (bf16_t)f.y;
    d[2*LDK] = (bf16_t)f.z;
    d[3*LDK] = (bf16_t)f.w;
  }
  __syncthreads();

  const int m0 = wv * 16;
  const int mA = lane & 15;         // A-fragment row
  const int kb = (lane >> 4) * 8;   // K sub-offset (0 or 8); other half at +16
  for (int nt = 0; nt < P2/16; ++nt) {
    const int n0 = nt * 16;
    f32x8 acc = {};
    const bf16_t* arow = sU   + (size_t)(m0 + mA)*LDK + kb;
    const bf16_t* brow = Bbar + (size_t)(n0 + (lane & 15))*HDIM + kb;
    for (int kt = 0; kt < HDIM/32; ++kt) {
      ABFrag a, bb;
      a.h[0]  = *(const bfx8*)(arow);
      a.h[1]  = *(const bfx8*)(arow + 16);
      bb.h[0] = *(const bfx8*)(brow);
      bb.h[1] = *(const bfx8*)(brow + 16);
      __builtin_prefetch((const void*)(brow + 64), 0, 1);
      acc = __builtin_amdgcn_wmma_f32_16x16x32_bf16(false, a.v, false, bb.v,
                                                    (short)0, acc, false, false);
      arow += 32; brow += 32;
    }
    const int n  = lane & 15;
    const int mb = (lane >> 4) * 8;  // D layout: VGPR r -> M = r (+8 for hi lanes)
    float* dst = Bu + ((size_t)b*LSEQ + l0 + m0 + mb)*P2 + n0 + n;
#pragma unroll
    for (int r = 0; r < 8; ++r) dst[(size_t)r*P2] = acc[r];
  }
}

// ---------------------------------------------------------------------------
// K2a: per-chunk local scan (x_t = lam*x_{t-1} + Bu_t), in place; save chunk end
// ---------------------------------------------------------------------------
__global__ void s5_scan_local(float* __restrict__ Bu, const float* __restrict__ lam_r,
                              const float* __restrict__ lam_i, float* __restrict__ carryA) {
  const int p = threadIdx.x, c = blockIdx.x, b = blockIdx.y;
  const float lr = lam_r[p], li = lam_i[p];
  float xr = 0.f, xi = 0.f;
  size_t base = ((size_t)b*LSEQ + (size_t)c*SCL) * P2 + p;
  for (int i = 0; i < SCL; ++i) {
    const float br = Bu[base], bi = Bu[base + PDIM];
    const float nr = lr*xr - li*xi + br;
    const float ni = lr*xi + li*xr + bi;
    xr = nr; xi = ni;
    Bu[base] = xr; Bu[base + PDIM] = xi;
    base += P2;
  }
  const size_t ci = (((size_t)b*NCH + c)*PDIM + p)*2;
  carryA[ci] = xr; carryA[ci+1] = xi;
}

// ---------------------------------------------------------------------------
// K2b: scan of chunk carries: entry(c) = lam^SCL * entry(c-1) + local_end(c-1)
// ---------------------------------------------------------------------------
__global__ void s5_scan_carry(const float* __restrict__ carryA, float* __restrict__ carryB,
                              const float* __restrict__ dscl_r, const float* __restrict__ dscl_i) {
  const int p = threadIdx.x, b = blockIdx.x;
  const float dr = dscl_r[p], di = dscl_i[p];
  float xr = 0.f, xi = 0.f;
  for (int c = 0; c < NCH; ++c) {
    const size_t ci = (((size_t)b*NCH + c)*PDIM + p)*2;
    carryB[ci] = xr; carryB[ci+1] = xi;
    const float ar = carryA[ci], ai = carryA[ci+1];
    const float nr = dr*xr - di*xi + ar;
    const float ni = dr*xi + di*xr + ai;
    xr = nr; xi = ni;
  }
}

// ---------------------------------------------------------------------------
// K2c: x_global = local + lam^(i+1)*carry; emit Xs bf16 = [2*re | -2*im]
// ---------------------------------------------------------------------------
__global__ void s5_scan_fixup(const float* __restrict__ Bu, const float* __restrict__ carryB,
                              const float* __restrict__ lam_r, const float* __restrict__ lam_i,
                              unsigned short* __restrict__ Xs_u16) {
  bf16_t* Xs = (bf16_t*)Xs_u16;
  const int p = threadIdx.x, c = blockIdx.x, b = blockIdx.y;
  const size_t ci = (((size_t)b*NCH + c)*PDIM + p)*2;
  const float cr = carryB[ci], cim = carryB[ci+1];
  const float lr = lam_r[p], li = lam_i[p];
  float pwr = lr, pwi = li;                         // lam^(i+1)
  size_t base = ((size_t)b*LSEQ + (size_t)c*SCL) * P2 + p;
  for (int i = 0; i < SCL; ++i) {
    const float xr = Bu[base]        + pwr*cr  - pwi*cim;
    const float xi = Bu[base + PDIM] + pwr*cim + pwi*cr;
    Xs[base]        = (bf16_t)( 2.f*xr);
    Xs[base + PDIM] = (bf16_t)(-2.f*xi);
    const float nr = pwr*lr - pwi*li, ni = pwr*li + pwi*lr;
    pwr = nr; pwi = ni;
    base += P2;
  }
}

// ---------------------------------------------------------------------------
// K3: ys[l][h] = sum_k Xs[l][k]*Ccat[h][k] + D[h]*u ; exact GELU; out [b][h][l]
// ---------------------------------------------------------------------------
__global__ void __launch_bounds__(256) s5_gemm_out(const unsigned short* __restrict__ Xs_u16,
    const unsigned short* __restrict__ Ccat_u16, const float* __restrict__ u,
    const float* __restrict__ Dvec, float* __restrict__ out) {
  extern __shared__ __align__(16) unsigned short smem_raw2[];
  bf16_t* sA = (bf16_t*)smem_raw2;                 // [CHUNK_M][P2+8]
  const int LDK = P2 + 8;
  const bf16_t* Xs   = (const bf16_t*)Xs_u16;
  const bf16_t* Ccat = (const bf16_t*)Ccat_u16;
  const int tid = threadIdx.x, lane = tid & 31, wv = tid >> 5;
  const int b = blockIdx.y;
  const int l0 = blockIdx.x * CHUNK_M;

  for (int r = wv; r < CHUNK_M; r += 8) {
    const bfx8* src = (const bfx8*)(Xs + ((size_t)b*LSEQ + l0 + r)*P2);
    bfx8* dst = (bfx8*)(sA + (size_t)r*LDK);
    for (int j = lane; j < P2/8; j += 32) dst[j] = src[j];
  }
  __syncthreads();

  const int m0 = wv * 16;
  const int mA = lane & 15;
  const int kb = (lane >> 4) * 8;
  for (int nt = 0; nt < HDIM/16; ++nt) {
    const int h0 = nt * 16;
    f32x8 acc = {};
    const bf16_t* arow = sA   + (size_t)(m0 + mA)*LDK + kb;
    const bf16_t* brow = Ccat + (size_t)(h0 + (lane & 15))*P2 + kb;
    for (int kt = 0; kt < P2/32; ++kt) {
      ABFrag a, bb;
      a.h[0]  = *(const bfx8*)(arow);
      a.h[1]  = *(const bfx8*)(arow + 16);
      bb.h[0] = *(const bfx8*)(brow);
      bb.h[1] = *(const bfx8*)(brow + 16);
      __builtin_prefetch((const void*)(brow + 64), 0, 1);
      acc = __builtin_amdgcn_wmma_f32_16x16x32_bf16(false, a.v, false, bb.v,
                                                    (short)0, acc, false, false);
      arow += 32; brow += 32;
    }
    const int n  = lane & 15;
    const int mb = (lane >> 4) * 8;
    const int h  = h0 + n;
    const float dh = Dvec[h];
#pragma unroll
    for (int r = 0; r < 8; ++r) {
      const int l = l0 + m0 + mb + r;
      const float uval = u[((size_t)b*HDIM + h)*LSEQ + l];
      const float y = acc[r] + dh * uval;
      const float g = 0.5f * y * (1.f + erff(y * 0.70710678118f));
      out[((size_t)b*HDIM + h)*LSEQ + l] = g;
    }
  }
}

// ---------------------------------------------------------------------------
extern "C" void kernel_launch(void* const* d_in, const int* in_sizes, int n_in,
                              void* d_out, int out_size, void* d_ws, size_t ws_size,
                              hipStream_t stream) {
  (void)in_sizes; (void)n_in; (void)out_size; (void)ws_size;
  const float* u        = (const float*)d_in[0];
  // d_in[1] = hiddens (unused)
  const float* Lre      = (const float*)d_in[2];
  const float* Lim      = (const float*)d_in[3];
  const float* Bmat     = (const float*)d_in[4];
  const float* Cmat     = (const float*)d_in[5];
  const float* Dvec     = (const float*)d_in[6];
  const float* log_step = (const float*)d_in[7];
  float* out = (float*)d_out;

  auto al = [](size_t x){ return (x + 255) & ~(size_t)255; };
  char* w = (char*)d_ws;
  unsigned short* Bbar = (unsigned short*)w; w += al((size_t)P2*HDIM*2);          // 2 MB
  unsigned short* Ccat = (unsigned short*)w; w += al((size_t)HDIM*P2*2);          // 2 MB
  float* lam_r  = (float*)w; w += al((size_t)PDIM*4);
  float* lam_i  = (float*)w; w += al((size_t)PDIM*4);
  float* dscl_r = (float*)w; w += al((size_t)PDIM*4);
  float* dscl_i = (float*)w; w += al((size_t)PDIM*4);
  float* carryA = (float*)w; w += al((size_t)BSZd*NCH*PDIM*2*4);                  // 256 KB
  float* carryB = (float*)w; w += al((size_t)BSZd*NCH*PDIM*2*4);                  // 256 KB
  float* Bu     = (float*)w; w += al((size_t)BSZd*LSEQ*P2*4);                     // 64 MB
  unsigned short* Xs = (unsigned short*)w; w += al((size_t)BSZd*LSEQ*P2*2);       // 32 MB

  const size_t smem1 = (size_t)CHUNK_M*(HDIM+8)*2;   // 258 KB (fits 320 KB WGP LDS)
  const size_t smem2 = (size_t)CHUNK_M*(P2+8)*2;     // 258 KB
  hipFuncSetAttribute((const void*)s5_gemm_bu,  hipFuncAttributeMaxDynamicSharedMemorySize, (int)smem1);
  hipFuncSetAttribute((const void*)s5_gemm_out, hipFuncAttributeMaxDynamicSharedMemorySize, (int)smem2);

  s5_prep_params<<<1, 512, 0, stream>>>(Lre, Lim, Bmat, log_step, Bbar,
                                        lam_r, lam_i, dscl_r, dscl_i);
  s5_prep_c<<<(HDIM*PDIM)/256, 256, 0, stream>>>(Cmat, Ccat);
  s5_gemm_bu<<<dim3(LSEQ/CHUNK_M, BSZd), 256, smem1, stream>>>(u, Bbar, Bu);
  s5_scan_local<<<dim3(NCH, BSZd), PDIM, 0, stream>>>(Bu, lam_r, lam_i, carryA);
  s5_scan_carry<<<BSZd, PDIM, 0, stream>>>(carryA, carryB, dscl_r, dscl_i);
  s5_scan_fixup<<<dim3(NCH, BSZd), PDIM, 0, stream>>>(Bu, carryB, lam_r, lam_i, Xs);
  s5_gemm_out<<<dim3(LSEQ/CHUNK_M, BSZd), 256, smem2, stream>>>(Xs, Ccat, u, Dvec, out);
}
